// TimeSeriesInteractionNetwork_61297773248570
// MI455X (gfx1250) — matrix operand order repaired and checked by reading, not compiled, not run on hardware
//
#include <hip/hip_runtime.h>
#include <hip/hip_bf16.h>
#include <math.h>

// ---------------------------------------------------------------------------
// Problem constants (from reference): B=8 T=2048 NB=8 CD=64 HD=256 H=2 OD=64
// ---------------------------------------------------------------------------
#define NT   16384      // B*T tokens
#define C_IN 512        // NB*CD
#define OD   64

typedef __attribute__((ext_vector_type(16))) __bf16 v16bf;
typedef __attribute__((ext_vector_type(8)))  float  v8f;

#define DEVI __device__ __forceinline__

// ---------------------------------------------------------------------------
// Tensor Data Mover support (gfx1250). This toolchain: 6-arg builtin
//   (uint32x4 g0, int32x8 g1, int32x4 g2, int32x4 g3, int32x8 g4, i32 cpol)
// D# packing per CDNA5 ISA ch.8: group0 = {flags, lds_addr, gaddr_lo, gaddr_hi|type},
// group1 = dims/strides. Groups 2/3 zero (<=2D tensor).
// ---------------------------------------------------------------------------
#if defined(__has_builtin)
#if __has_builtin(__builtin_amdgcn_tensor_load_to_lds) && \
    __has_builtin(__builtin_amdgcn_s_wait_tensorcnt)
#define HAVE_TDM 1
#endif
#endif
#ifndef HAVE_TDM
#define HAVE_TDM 0
#endif

#if HAVE_TDM
typedef unsigned int v4ui __attribute__((ext_vector_type(4)));
typedef int          v4i_ __attribute__((ext_vector_type(4)));
typedef int          v8i_ __attribute__((ext_vector_type(8)));

// Copy rows x width bf16 elements: global (row stride = stride elems) -> LDS
// (contiguous). Issued once per calling wave; caller waits tensorcnt.
DEVI void tdm_load_2d_bf16(unsigned lds_off, const __bf16* gptr,
                           unsigned width, unsigned rows, unsigned stride) {
  unsigned long long ga = (unsigned long long)(uintptr_t)gptr;
  v4ui g0;
  g0.x = 1u;                                   // count=1 (valid user D#)
  g0.y = lds_off;                              // lds_addr, bytes
  g0.z = (unsigned)(ga & 0xffffffffu);         // global_addr[31:0]
  g0.w = (unsigned)((ga >> 32) & 0x01ffffffu)  // global_addr[56:32]
       | (2u << 30);                           // type = 2 ("image")
  v8i_ g1;
  g1[0] = (int)(1u << 16);                              // data_size=1 -> 2 bytes
  g1[1] = (int)((width & 0xffffu) << 16);               // tensor_dim0[15:0]
  g1[2] = (int)(((width >> 16) & 0xffffu)               // tensor_dim0[31:16]
             | ((rows & 0xffffu) << 16));               // tensor_dim1[15:0]
  g1[3] = (int)(((rows >> 16) & 0xffffu)                // tensor_dim1[31:16]
             | ((width & 0xffffu) << 16));              // tile_dim0
  g1[4] = (int)(rows & 0xffffu);                        // tile_dim1 (tile_dim2=0)
  g1[5] = (int)stride;                                  // tensor_dim0_stride[31:0]
  g1[6] = 0;                                            // stride hi / dim1_stride lo
  g1[7] = 0;
  v4i_ z4 = {0, 0, 0, 0};
  v8i_ z8 = {0, 0, 0, 0, 0, 0, 0, 0};
  __builtin_amdgcn_tensor_load_to_lds(g0, g1, z4, z4, z8, 0);
}
#endif

// Stage a [rows x width] bf16 panel (global row stride = stride) into
// contiguous LDS. Includes the trailing workgroup barrier.
DEVI void stage_load(__bf16* dst, const __bf16* src, unsigned width,
                     unsigned rows, unsigned stride, int tid) {
#if HAVE_TDM
  if ((tid >> 5) == 0) {
    tdm_load_2d_bf16((unsigned)(uintptr_t)dst, src, width, rows, stride);
    __builtin_amdgcn_s_wait_tensorcnt((short)0);
  }
#else
  for (unsigned i = tid; i < width * rows; i += 256) {
    unsigned r = i / width, c = i % width;
    dst[i] = src[(size_t)r * stride + c];
  }
#endif
  __syncthreads();
}

// ---------------------------------------------------------------------------
// WMMA helpers (bf16 x bf16 -> f32, 16x16x32)
// ---------------------------------------------------------------------------
DEVI v8f wmma_bf16(v16bf a, v16bf b, v8f c) {
  return __builtin_amdgcn_wmma_f32_16x16x32_bf16(
      false, a, false, b, (short)0, c, false, false);
}

// A fragment: 16x32 tile, row-major LDS, leading dim = ld (elements).
DEVI v16bf frag_a(const __bf16* p, int ld, int lane) {
  int m  = lane & 15;
  int kb = (lane & 16) ? 8 : 0;
  v16bf r;
#pragma unroll
  for (int q = 0; q < 8; ++q) {
    int k = ((q >> 2) << 4) + kb + ((q & 3) << 1);
    r[2 * q]     = p[m * ld + k];
    r[2 * q + 1] = p[m * ld + k + 1];
  }
  return r;
}

// B fragment: 32x16 tile, row-major [K][N] LDS; N from lane, K from element.
DEVI v16bf frag_b(const __bf16* p, int ld, int lane) {
  int n  = lane & 15;
  int kb = (lane & 16) ? 8 : 0;
  v16bf r;
#pragma unroll
  for (int i = 0; i < 16; ++i) {
    int k = (i < 8) ? (kb + i) : (16 + kb + (i - 8));
    r[i] = p[k * ld + n];
  }
  return r;
}

DEVI void store_c(float* C, int ld, int lane, int m0, int n0, v8f c) {
  int n  = n0 + (lane & 15);
  int mb = (lane & 16) ? 8 : 0;
#pragma unroll
  for (int r = 0; r < 8; ++r) C[(m0 + mb + r) * ld + n] = c[r];
}

// C[M][N](f32,LDS) = A[M][K](bf16,LDS) * B[K][N](bf16,LDS)
DEVI void gemm_lds(const __bf16* A, const __bf16* B, float* C,
                   int M, int N, int K, int lane, int wave, int nwaves) {
  int mt = M >> 4, nt = N >> 4, kt = K >> 5;
  for (int t = wave; t < mt * nt; t += nwaves) {
    int tm = t / nt, tn = t % nt;
    v8f acc = {};
    for (int kk = 0; kk < kt; ++kk)
      acc = wmma_bf16(frag_a(A + (tm * 16) * K + kk * 32, K, lane),
                      frag_b(B + (kk * 32) * N + tn * 16, N, lane), acc);
    store_c(C, N, lane, tm * 16, tn * 16, acc);
  }
}

DEVI float gelu_exact(float x) {
  return 0.5f * x * (1.0f + erff(x * 0.70710678118654752f));
}

// 64 rows x 256 cols LayerNorm stats with 256 threads (4 threads / row).
DEVI void ln_stats64(const float* A, float* pS, float* pS2,
                     float* rowM, float* rowIV, int tid) {
  int r = tid >> 2, q = tid & 3;
  float s = 0.f, s2 = 0.f;
  for (int c = q * 64; c < q * 64 + 64; ++c) {
    float v = A[r * 256 + c]; s += v; s2 += v * v;
  }
  pS[r * 4 + q] = s; pS2[r * 4 + q] = s2;
  __syncthreads();
  if (tid < 64) {
    float S  = pS[tid * 4] + pS[tid * 4 + 1] + pS[tid * 4 + 2] + pS[tid * 4 + 3];
    float S2 = pS2[tid * 4] + pS2[tid * 4 + 1] + pS2[tid * 4 + 2] + pS2[tid * 4 + 3];
    float m = S * (1.f / 256.f);
    float v = S2 * (1.f / 256.f) - m * m;
    rowM[tid]  = m;
    rowIV[tid] = rsqrtf(v + 1e-5f);
  }
  __syncthreads();
}

// ---------------------------------------------------------------------------
// K0: f32 -> bf16 pre-pack (convert once, DMA many times)
// ---------------------------------------------------------------------------
__global__ __launch_bounds__(256)
void k_pack(const float* __restrict__ src, __bf16* __restrict__ dst, int n) {
  for (int i = blockIdx.x * 256 + threadIdx.x; i < n; i += gridDim.x * 256)
    dst[i] = (__bf16)src[i];
}

// ---------------------------------------------------------------------------
// K1: per-block MLP   proc[t][n][c]  (32 tokens x one block per WG)
// LDS: w1 32K | w2 32K | xs 4K | hf 32K | hb 16K | st ~2.3K  => 121088 B
// ---------------------------------------------------------------------------
#define K1_SHMEM 121088
__global__ __launch_bounds__(256)
void k_block_mlp(const __bf16* __restrict__ xp, const __bf16* __restrict__ W1p,
                 const __bf16* __restrict__ W2p,
                 const float* __restrict__ b1, const float* __restrict__ g,
                 const float* __restrict__ be, const float* __restrict__ b2,
                 float* __restrict__ proc) {
  extern __shared__ char smem[];
  __bf16* w1 = (__bf16*)smem;                 // 64x256
  __bf16* w2 = (__bf16*)(smem + 32768);       // 256x64
  __bf16* xs = (__bf16*)(smem + 65536);       // 32x64
  float*  hf = (float*)(smem + 69632);        // 32x256
  __bf16* hb = (__bf16*)(smem + 102400);      // 32x256
  float*  st = (float*)(smem + 118784);       // [256]+[256]+[64]

  int tid = threadIdx.x, lane = tid & 31, wave = tid >> 5;
  int n  = blockIdx.y;
  int t0 = blockIdx.x * 32;

#if HAVE_TDM
  if (wave == 0) {
    tdm_load_2d_bf16((unsigned)(uintptr_t)w1, W1p + (size_t)n * 64 * 256,
                     16384, 1, 16384);
    tdm_load_2d_bf16((unsigned)(uintptr_t)w2, W2p + (size_t)n * 256 * 64,
                     16384, 1, 16384);
    tdm_load_2d_bf16((unsigned)(uintptr_t)xs, xp + (size_t)t0 * C_IN + n * 64,
                     64, 32, C_IN);
    __builtin_amdgcn_s_wait_tensorcnt((short)0);
  }
#else
  for (int i = tid; i < 64 * 256; i += 256) w1[i] = W1p[(size_t)n * 64 * 256 + i];
  for (int i = tid; i < 256 * 64; i += 256) w2[i] = W2p[(size_t)n * 256 * 64 + i];
  for (int i = tid; i < 32 * 64; i += 256) {
    int tt = i >> 6, c = i & 63;
    xs[i] = xp[(size_t)(t0 + tt) * C_IN + n * 64 + c];
  }
#endif
  __syncthreads();

  gemm_lds(xs, w1, hf, 32, 256, 64, lane, wave, 8);     // h = x @ W1
  __syncthreads();
  for (int i = tid; i < 32 * 256; i += 256) hf[i] += b1[n * 256 + (i & 255)];
  __syncthreads();

  {
    int r = tid >> 3, ch = tid & 7;
    float s = 0.f, s2 = 0.f;
    for (int c = ch * 32; c < ch * 32 + 32; ++c) {
      float v = hf[r * 256 + c]; s += v; s2 += v * v;
    }
    st[r * 8 + ch] = s; st[256 + r * 8 + ch] = s2;
  }
  __syncthreads();
  if (tid < 32) {
    float s = 0.f, s2 = 0.f;
    for (int ch = 0; ch < 8; ++ch) { s += st[tid * 8 + ch]; s2 += st[256 + tid * 8 + ch]; }
    float m = s * (1.f / 256.f), v = s2 * (1.f / 256.f) - m * m;
    st[512 + tid * 2] = m; st[512 + tid * 2 + 1] = rsqrtf(v + 1e-5f);
  }
  __syncthreads();
  for (int i = tid; i < 32 * 256; i += 256) {
    int r = i >> 8, c = i & 255;
    float y = (hf[i] - st[512 + r * 2]) * st[512 + r * 2 + 1] * g[n * 256 + c] + be[n * 256 + c];
    hb[i] = (__bf16)gelu_exact(y);
  }
  __syncthreads();

  gemm_lds(hb, w2, hf, 32, 64, 256, lane, wave, 8);     // proc = h @ W2
  __syncthreads();
  for (int i = tid; i < 32 * 64; i += 256) {
    int tt = i >> 6, c = i & 63;
    proc[(size_t)(t0 + tt) * C_IN + n * 64 + c] = hf[i] + b2[n * 64 + c];
  }
}

// ---------------------------------------------------------------------------
// K2: edge pipeline.  4 tokens/WG -> cat rows M=64, seq of WMMA GEMMs with
// TDM-staged bf16 weight panels reusing one LDS weight buffer.
// ---------------------------------------------------------------------------
#define K2_SHMEM 183040
__global__ __launch_bounds__(256)
void k_edge(const float* __restrict__ proc,
            const __bf16* __restrict__ gW1p, const __bf16* __restrict__ iW1p,
            const __bf16* __restrict__ iW2p, const __bf16* __restrict__ aWp,
            const __bf16* __restrict__ aWop,
            const float* __restrict__ gb1, const float* __restrict__ gg,
            const float* __restrict__ gbe, const float* __restrict__ gW2,
            const float* __restrict__ gb2,
            const float* __restrict__ ib1, const float* __restrict__ ig,
            const float* __restrict__ ibe, const float* __restrict__ ib2,
            const float* __restrict__ ab,  const float* __restrict__ abo,
            const float* __restrict__ edge_w, float* __restrict__ weighted) {
  extern __shared__ char smem[];
  __bf16* catB   = (__bf16*)smem;                  // [64][128]
  __bf16* wB     = (__bf16*)(smem + 16384);        // weight panel (<=64KB)
  float*  actF   = (float*)(smem + 81920);         // [64][256] f32 staging
  __bf16* i1B    = (__bf16*)(smem + 147456);       // [64][256]
  __bf16* interB = (__bf16*)smem;                  // alias catB after stage 3
  __bf16* aB     = (__bf16*)(smem + 147456);       // alias i1B after stage 4
  float*  gateF  = (float*)(smem + 180224);        // [64]
  float*  pS     = (float*)(smem + 180480);        // [64][4]
  float*  pS2    = pS + 256;
  float*  rowM   = pS2 + 256;
  float*  rowIV  = rowM + 64;

  int tid = threadIdx.x, lane = tid & 31, wave = tid >> 5;
  int t0 = blockIdx.x * 4;

  // Build cat: row = tt*16 + e*2 + s ; s=0:[pu,pv] s=1:[pv,pu]; pu=e pv=(e+1)&7
  for (int i = tid; i < 64 * 128; i += 256) {
    int row = i >> 7, c = i & 127;
    int tt = row >> 4, es = row & 15, e = es >> 1, s = es & 1;
    int half = c >> 6, cc = c & 63;
    int blk = (half ^ s) ? ((e + 1) & 7) : e;
    catB[i] = (__bf16)proc[(size_t)(t0 + tt) * C_IN + blk * 64 + cc];
  }

  // -------- gate branch --------
  stage_load(wB, gW1p, 128 * 256, 1, 128 * 256, tid);
  gemm_lds(catB, wB, actF, 64, 256, 128, lane, wave, 8);
  __syncthreads();
  for (int i = tid; i < 64 * 256; i += 256) actF[i] += gb1[i & 255];
  __syncthreads();
  ln_stats64(actF, pS, pS2, rowM, rowIV, tid);
  {
    int r = tid >> 2, q = tid & 3;
    float acc = 0.f;
    for (int c = q * 64; c < q * 64 + 64; ++c) {
      float y = (actF[r * 256 + c] - rowM[r]) * rowIV[r] * gg[c] + gbe[c];
      acc += gelu_exact(y) * gW2[c];
    }
    pS[r * 4 + q] = acc;
  }
  __syncthreads();
  if (tid < 64) {
    float d = pS[tid * 4] + pS[tid * 4 + 1] + pS[tid * 4 + 2] + pS[tid * 4 + 3] + gb2[0];
    gateF[tid] = 1.f / (1.f + expf(-d));
  }
  __syncthreads();

  // -------- int branch --------
  stage_load(wB, iW1p, 128 * 256, 1, 128 * 256, tid);
  gemm_lds(catB, wB, actF, 64, 256, 128, lane, wave, 8);
  __syncthreads();
  for (int i = tid; i < 64 * 256; i += 256) actF[i] += ib1[i & 255];
  __syncthreads();
  ln_stats64(actF, pS, pS2, rowM, rowIV, tid);
  for (int i = tid; i < 64 * 256; i += 256) {
    int r = i >> 8, c = i & 255;
    float y = (actF[i] - rowM[r]) * rowIV[r] * ig[c] + ibe[c];
    y = gelu_exact(y);
    i1B[i] = (__bf16)(y > 0.f ? y : 0.f);
  }
  __syncthreads();

  // inter = i1 @ iW2 (256x128) + ib2
  stage_load(wB, iW2p, 256 * 128, 1, 256 * 128, tid);
  gemm_lds(i1B, wB, actF, 64, 128, 256, lane, wave, 8);
  __syncthreads();
  for (int i = tid; i < 64 * 128; i += 256) interB[i] = (__bf16)(actF[i] + ib2[i & 127]);
  __syncthreads();

  // a = inter @ attn_W (128x128) + ab
  stage_load(wB, aWp, 128 * 128, 1, 128 * 128, tid);
  gemm_lds(interB, wB, actF, 64, 128, 128, lane, wave, 8);
  __syncthreads();
  for (int i = tid; i < 64 * 128; i += 256) aB[i] = (__bf16)(actF[i] + ab[i & 127]);
  __syncthreads();

  // a2 = a @ attn_Wo (128x64) + abo ; weighted = sum_s a2*gate*edge_w
  stage_load(wB, aWop, 128 * 64, 1, 128 * 64, tid);
  gemm_lds(aB, wB, actF, 64, 64, 128, lane, wave, 8);
  __syncthreads();
  for (int i = tid; i < 4 * 8 * 64; i += 256) {
    int tt = i >> 9, rem = i & 511, e = rem >> 6, c = rem & 63;
    int r0 = tt * 16 + e * 2, r1 = r0 + 1;
    float w = (actF[r0 * 64 + c] + abo[c]) * gateF[r0] * edge_w[e * 2 + 0]
            + (actF[r1 * 64 + c] + abo[c]) * gateF[r1] * edge_w[e * 2 + 1];
    weighted[(size_t)(t0 + tt) * C_IN + e * 64 + c] = w;
  }
}

// ---------------------------------------------------------------------------
// K3: out = proc + scatter(weighted); red = einsum(out, out_W) + out_b
// ---------------------------------------------------------------------------
__global__ __launch_bounds__(64)
void k_reduce(const float* __restrict__ proc, const float* __restrict__ weighted,
              const float* __restrict__ out_W, const float* __restrict__ out_b,
              float* __restrict__ red) {
  __shared__ float ot[8][64];
  int t = blockIdx.x, tid = threadIdx.x;
  for (int n = 0; n < 8; ++n) {
    size_t base = (size_t)t * C_IN + n * 64 + tid;
    ot[n][tid] = proc[base] + weighted[base]
               + weighted[(size_t)t * C_IN + (((n + 7) & 7)) * 64 + tid];
  }
  __syncthreads();
  int n = tid >> 3, o = tid & 7;
  float acc = out_b[n * 8 + o];
  for (int c = 0; c < 64; ++c) acc += ot[n][c] * out_W[n * 512 + c * 8 + o];
  red[(size_t)t * OD + tid] = acc;
}

// ---------------------------------------------------------------------------
// K4: flash attention per (b, h, 64-query tile); head_dim=32, T=2048
// ---------------------------------------------------------------------------
__global__ __launch_bounds__(256)
void k_attn(const float* __restrict__ red, const float* __restrict__ tau,
            const float* __restrict__ delta, float* __restrict__ attnV) {
  __shared__ __bf16 qB[64 * 32];
  __shared__ __bf16 kT[32 * 64];     // K^T  [d][j]
  __shared__ __bf16 vB[64 * 32];     // V    [j][d]
  __shared__ float  sF[64 * 64];
  __shared__ __bf16 pB[64 * 64];
  __shared__ float  rowMax[64], rowSum[64], rowScale[64];
  __shared__ float  pMax[64 * 4], pSum[64 * 4];

  int tid = threadIdx.x, lane = tid & 31, wave = tid >> 5;
  int qt = blockIdx.x & 31;
  int h  = (blockIdx.x >> 5) & 1;
  int b  = blockIdx.x >> 6;
  float stau   = rsqrtf(32.f) * tau[0];
  float sdelta = rsqrtf(32.f) * delta[0];
  size_t base = (size_t)b * 2048 * OD + h * 32;

  for (int i = tid; i < 64 * 32; i += 256) {
    int r = i >> 5, d = i & 31;
    qB[i] = (__bf16)red[base + (size_t)(qt * 64 + r) * OD + d];
  }
  v8f o_acc = {};
  if (tid < 64) { rowMax[tid] = -1e30f; rowSum[tid] = 0.f; }
  __syncthreads();

  for (int kc = 0; kc < 32; ++kc) {
    for (int i = tid; i < 64 * 32; i += 256) {
      int r = i >> 5, d = i & 31;
      float v = red[base + (size_t)(kc * 64 + r) * OD + d];
      kT[d * 64 + r] = (__bf16)v;
      vB[i]          = (__bf16)v;
    }
    __syncthreads();

    for (int t2 = wave * 2; t2 < wave * 2 + 2; ++t2) {
      int tm = t2 >> 2, tn = t2 & 3;
      v8f acc = {};
      acc = wmma_bf16(frag_a(qB + tm * 16 * 32, 32, lane),
                      frag_b(kT + tn * 16, 64, lane), acc);
      int n = tn * 16 + (lane & 15), mb = (lane & 16) ? 8 : 0;
#pragma unroll
      for (int r = 0; r < 8; ++r)
        sF[(tm * 16 + mb + r) * 64 + n] = acc[r] * stau + sdelta;
    }
    __syncthreads();

    {
      int r = tid >> 2, q = tid & 3;
      float mx = -1e30f;
      for (int c = q * 16; c < q * 16 + 16; ++c) mx = fmaxf(mx, sF[r * 64 + c]);
      pMax[r * 4 + q] = mx;
    }
    __syncthreads();
    if (tid < 64) {
      float mc = fmaxf(fmaxf(pMax[tid * 4], pMax[tid * 4 + 1]),
                       fmaxf(pMax[tid * 4 + 2], pMax[tid * 4 + 3]));
      float nm = fmaxf(rowMax[tid], mc);
      rowScale[tid] = expf(rowMax[tid] - nm);
      rowMax[tid] = nm;
    }
    __syncthreads();
    {
      int r = tid >> 2, q = tid & 3;
      float s = 0.f, nm = rowMax[r];
      for (int c = q * 16; c < q * 16 + 16; ++c) {
        float p = expf(sF[r * 64 + c] - nm);
        pB[r * 64 + c] = (__bf16)p;
        s += p;
      }
      pSum[r * 4 + q] = s;
    }
    __syncthreads();
    if (tid < 64)
      rowSum[tid] = rowSum[tid] * rowScale[tid]
                  + pSum[tid * 4] + pSum[tid * 4 + 1] + pSum[tid * 4 + 2] + pSum[tid * 4 + 3];
    __syncthreads();

    {
      int mt = wave >> 1, nt = wave & 1;
      int mb = (lane & 16) ? 8 : 0;
#pragma unroll
      for (int r = 0; r < 8; ++r) o_acc[r] *= rowScale[mt * 16 + mb + r];
      for (int kk = 0; kk < 2; ++kk)
        o_acc = wmma_bf16(frag_a(pB + mt * 16 * 64 + kk * 32, 64, lane),
                          frag_b(vB + kk * 32 * 32 + nt * 16, 32, lane), o_acc);
    }
    __syncthreads();
  }

  {
    int mt = wave >> 1, nt = wave & 1;
    int n = nt * 16 + (lane & 15), mb = (lane & 16) ? 8 : 0;
#pragma unroll
    for (int r = 0; r < 8; ++r) {
      int row = mt * 16 + mb + r;
      attnV[base + (size_t)(qt * 64 + row) * OD + n] = o_acc[r] / rowSum[row];
    }
  }
}

// ---------------------------------------------------------------------------
// K5: output = attnV @ glob_W + glob_b ; also l1 scalar
// ---------------------------------------------------------------------------
__global__ __launch_bounds__(64)
void k_out(const float* __restrict__ attnV, const float* __restrict__ gW,
           const float* __restrict__ gb, const float* __restrict__ edge_w,
           float* __restrict__ out) {
  __shared__ float v[64];
  int t = blockIdx.x, tid = threadIdx.x;
  v[tid] = attnV[(size_t)t * OD + tid];
  __syncthreads();
  float acc = gb[tid];
  for (int c = 0; c < 64; ++c) acc += v[c] * gW[c * 64 + tid];
  out[(size_t)t * OD + tid] = acc;
  if (t == 0 && tid == 0) {
    float l1 = 0.f;
    for (int i = 0; i < 16; ++i) l1 += fabsf(edge_w[i]);
    out[(size_t)NT * OD] = l1;
  }
}

// ---------------------------------------------------------------------------
extern "C" void kernel_launch(void* const* d_in, const int* in_sizes, int n_in,
                              void* d_out, int out_size, void* d_ws, size_t ws_size,
                              hipStream_t stream) {
  (void)in_sizes; (void)n_in; (void)out_size; (void)ws_size;
  const float* x      = (const float*)d_in[0];
  const float* blk_W1 = (const float*)d_in[1];
  const float* blk_b1 = (const float*)d_in[2];
  const float* blk_g  = (const float*)d_in[3];
  const float* blk_be = (const float*)d_in[4];
  const float* blk_W2 = (const float*)d_in[5];
  const float* blk_b2 = (const float*)d_in[6];
  const float* gW1    = (const float*)d_in[7];
  const float* gb1    = (const float*)d_in[8];
  const float* gg     = (const float*)d_in[9];
  const float* gbe    = (const float*)d_in[10];
  const float* gW2    = (const float*)d_in[11];
  const float* gb2    = (const float*)d_in[12];
  const float* iW1    = (const float*)d_in[13];
  const float* ib1    = (const float*)d_in[14];
  const float* ig     = (const float*)d_in[15];
  const float* ibe    = (const float*)d_in[16];
  const float* iW2    = (const float*)d_in[17];
  const float* ib2    = (const float*)d_in[18];
  const float* aW     = (const float*)d_in[19];
  const float* ab     = (const float*)d_in[20];
  const float* aWo    = (const float*)d_in[21];
  const float* abo    = (const float*)d_in[22];
  const float* out_W  = (const float*)d_in[23];
  const float* out_b  = (const float*)d_in[24];
  const float* edge_w = (const float*)d_in[25];
  const float* tau    = (const float*)d_in[26];
  const float* delta  = (const float*)d_in[27];
  const float* glob_W = (const float*)d_in[28];
  const float* glob_b = (const float*)d_in[29];
  float* out = (float*)d_out;

  // ---- workspace layout ----
  float* proc     = (float*)d_ws;                    // NT*512 f32
  float* weighted = proc + (size_t)NT * C_IN;        // NT*512 f32
  float* red      = weighted + (size_t)NT * C_IN;    // NT*64  f32
  float* attnV    = red + (size_t)NT * OD;           // NT*64  f32
  __bf16* xp      = (__bf16*)(attnV + (size_t)NT * OD);  // NT*512 bf16
  __bf16* wpW1    = xp + (size_t)NT * C_IN;          // 8*64*256
  __bf16* wpW2    = wpW1 + 8 * 64 * 256;             // 8*256*64
  __bf16* wpGW1   = wpW2 + 8 * 256 * 64;             // 128*256
  __bf16* wpIW1   = wpGW1 + 128 * 256;               // 128*256
  __bf16* wpIW2   = wpIW1 + 128 * 256;               // 256*128
  __bf16* wpAW    = wpIW2 + 256 * 128;               // 128*128
  __bf16* wpAWo   = wpAW + 128 * 128;                // 128*64

  // ---- K0: one-shot bf16 pre-pack ----
  k_pack<<<dim3(2048), dim3(256), 0, stream>>>(x, xp, NT * C_IN);
  k_pack<<<dim3(512),  dim3(256), 0, stream>>>(blk_W1, wpW1, 8 * 64 * 256);
  k_pack<<<dim3(512),  dim3(256), 0, stream>>>(blk_W2, wpW2, 8 * 256 * 64);
  k_pack<<<dim3(128),  dim3(256), 0, stream>>>(gW1, wpGW1, 128 * 256);
  k_pack<<<dim3(128),  dim3(256), 0, stream>>>(iW1, wpIW1, 128 * 256);
  k_pack<<<dim3(128),  dim3(256), 0, stream>>>(iW2, wpIW2, 256 * 128);
  k_pack<<<dim3(64),   dim3(256), 0, stream>>>(aW, wpAW, 128 * 128);
  k_pack<<<dim3(32),   dim3(256), 0, stream>>>(aWo, wpAWo, 128 * 64);

  k_block_mlp<<<dim3(NT / 32, 8), dim3(256), K1_SHMEM, stream>>>(
      xp, wpW1, wpW2, blk_b1, blk_g, blk_be, blk_b2, proc);

  k_edge<<<dim3(NT / 4), dim3(256), K2_SHMEM, stream>>>(
      proc, wpGW1, wpIW1, wpIW2, wpAW, wpAWo,
      gb1, gg, gbe, gW2, gb2, ib1, ig, ibe, ib2, ab, abo, edge_w, weighted);

  k_reduce<<<dim3(NT), dim3(64), 0, stream>>>(proc, weighted, out_W, out_b, red);

  k_attn<<<dim3(8 * 2 * 32), dim3(256), 0, stream>>>(red, tau, delta, attnV);

  k_out<<<dim3(NT), dim3(64), 0, stream>>>(attnV, glob_W, glob_b, edge_w, out);
}
